// GNN_43843026158107
// MI455X (gfx1250) — compile-verified, matrix operand order checked by reading, not run on hardware
//
#include <hip/hip_runtime.h>
#include <hip/hip_bf16.h>
#include <math.h>

// ---------------------------------------------------------------------------
// GNN forward for MI455X (gfx1250). Dense MLP blocks use f16 WMMA with f32
// accumulate (v_wmma_f32_16x16x32_f16); global->LDS staging uses the CDNA5
// async-to-LDS path when available; scatter/pool via float atomics.
// ---------------------------------------------------------------------------

typedef _Float16 half_t;
typedef __attribute__((ext_vector_type(16))) _Float16 v16h;
typedef __attribute__((ext_vector_type(8)))  float    v8f;

#define GHID 64
#define G_ALPHA 0.9f
#define G_LN_EPS 1e-5f
#define EG1_MAXK 192   // edge MLP1 max padded K (2*64+64)
#define NG1_MAXK 288   // node MLP1 max padded K (64+193 -> 288)

// ---- CDNA5 async global->LDS (GLOBAL_LOAD_ASYNC_TO_LDS_B128, ASYNCcnt) ----
// Probe-verified: builtin expects (v4i addrspace(1)*, v4i addrspace(3)*, imm, imm).
#if defined(__gfx1250__) && __has_builtin(__builtin_amdgcn_global_load_async_to_lds_b128)
  #define HAVE_ASYNC_LDS 1
  typedef __attribute__((ext_vector_type(4))) int v4i_t;
  typedef const __attribute__((address_space(1))) char gchar_t;
  typedef __attribute__((address_space(3))) char lchar_t;
  typedef __attribute__((address_space(1))) v4i_t gvec_t;
  typedef __attribute__((address_space(3))) v4i_t lvec_t;
  #define ASYNC_LDS_LOAD16(gp, lp) \
      __builtin_amdgcn_global_load_async_to_lds_b128((gvec_t*)(gp), (lvec_t*)(lp), 0, 0)
#else
  #define HAVE_ASYNC_LDS 0
#endif

__device__ __forceinline__ void async_lds_wait_all() {
#if HAVE_ASYNC_LDS
  #if __has_builtin(__builtin_amdgcn_s_wait_asynccnt)
    __builtin_amdgcn_s_wait_asynccnt(0);
  #else
    asm volatile("s_wait_asynccnt 0x0" ::: "memory");
  #endif
#endif
}

// Stage nhalfs f16 values global->LDS (16B granularity; nhalfs % 8 == 0).
__device__ __forceinline__ void stage_lds16(half_t* dst, const half_t* src,
                                            int nhalfs, int tid, int nthreads) {
#if HAVE_ASYNC_LDS
    gchar_t* g = (gchar_t*)src;
    lchar_t* l = (lchar_t*)dst;
    const int nchunks = nhalfs >> 3;
    for (int i = tid; i < nchunks; i += nthreads)
        ASYNC_LDS_LOAD16(g + (size_t)i * 16, l + (size_t)i * 16);
#else
    const uint4* s = (const uint4*)src;
    uint4* d = (uint4*)dst;
    const int nchunks = nhalfs >> 3;
    for (int i = tid; i < nchunks; i += nthreads) d[i] = s[i];
#endif
}

// Copy one 16B chunk global->LDS (addresses 16B aligned).
__device__ __forceinline__ void copy_chunk16(half_t* dst, const half_t* src) {
#if HAVE_ASYNC_LDS
    ASYNC_LDS_LOAD16((gchar_t*)src, (lchar_t*)dst);
#else
    *(uint4*)dst = *(const uint4*)src;
#endif
}

__device__ __forceinline__ float gelu_exact(float x) {
    return 0.5f * x * (1.0f + erff(x * 0.7071067811865476f));
}
__device__ __forceinline__ float softplus_f(float x) {
    return (x > 30.0f) ? x : log1pf(expf(x));
}
__device__ __forceinline__ void atomic_max_f32(float* addr, float v) {
    if (v >= 0.0f) atomicMax((int*)addr, __float_as_int(v));
    else           atomicMin((unsigned int*)addr, (unsigned int)__float_as_int(v));
}

// -------------------------- utility kernels --------------------------------

__global__ void k_fill_f32(float* p, long n, float v) {
    long i = (long)blockIdx.x * blockDim.x + threadIdx.x;
    if (i < n) p[i] = v;
}

// dst[r*ldd + c] = (half)src[r*cols + c]
__global__ void k_cvt_f16(const float* __restrict__ src, half_t* __restrict__ dst,
                          long rows, int cols, int ldd) {
    long i = (long)blockIdx.x * blockDim.x + threadIdx.x;
    if (i >= rows * (long)cols) return;
    long r = i / cols; int c = (int)(i % cols);
    dst[r * ldd + c] = (half_t)src[i];
}

__global__ void k_deg(const int* __restrict__ col, float* __restrict__ deg, int E) {
    int e = blockIdx.x * blockDim.x + threadIdx.x;
    if (e < E) atomicAdd(&deg[col[e]], 1.0f);
}

__global__ void k_gcnt(const int* __restrict__ batch, float* __restrict__ gcnt, int N) {
    int n = blockIdx.x * blockDim.x + threadIdx.x;
    if (n < N) atomicAdd(&gcnt[batch[n]], 1.0f);
}

// Pack W[K x 64] (row major, fp32) into per-lane WMMA B-fragment order:
// out[((kt*4+nt)*32 + lane)*16 + j],  k = kt*32 + (lane>>4)*16 + j,
// c = nt*16 + (lane&15).  Zero-padded for k >= K.
__global__ void k_pack_w(const float* __restrict__ W, half_t* __restrict__ out,
                         int K, int Kpad) {
    int idx = blockIdx.x * blockDim.x + threadIdx.x;
    if (idx >= Kpad * GHID) return;
    int j    = idx & 15;
    int lane = (idx >> 4) & 31;
    int t    = idx >> 9;          // (kt*4 + nt)
    int nt   = t & 3;
    int kt   = t >> 2;
    int k = kt * 32 + ((lane >> 4) << 4) + j;
    int c = nt * 16 + (lane & 15);
    out[idx] = (k < K) ? (half_t)W[k * GHID + c] : (half_t)0.0f;
}

// ------------------------- edge MLP layer 1 --------------------------------
// z1[e] = gelu( concat(h[row], h[col], eattr) @ W1 + b1 )   (E x 64, f16 out)
__global__ void __launch_bounds__(128)
k_edge_gemm1(int E, int nh, int ne, int Kpad,
             const half_t* __restrict__ hf16, const half_t* __restrict__ ef16,
             const int* __restrict__ erow, const int* __restrict__ ecol,
             const half_t* __restrict__ pw, const float* __restrict__ bias,
             half_t* __restrict__ z1) {
    __shared__ __align__(32) half_t At[4][16 * EG1_MAXK];
    __shared__ __align__(32) half_t Wt[EG1_MAXK * GHID];
    const int tid = threadIdx.x, wave = tid >> 5, lane = tid & 31;
    const int m = lane & 15, kh = lane >> 4;
    const int nk = Kpad >> 5;

    stage_lds16(Wt, pw, Kpad * GHID, tid, 128);

    const int tile = blockIdx.x * 4 + wave;
    // gather A tile: 16 rows x Kpad, each lane fills half a row
    {
        const int eid = tile * 16 + m;
        half_t* a = &At[wave][m * Kpad];
        const int khalf = Kpad >> 1;
        const int k0 = kh * khalf, k1 = k0 + khalf;
        if (eid < E) {
            const int r = erow[eid], c = ecol[eid];
            const half_t* hr = &hf16[(long)r * GHID];
            const half_t* hc = &hf16[(long)c * GHID];
            const half_t* ee = &ef16[(long)eid * GHID];
            if (((nh | ne) & 7) == 0) {
                // 16B-aligned segments: copy 8-half chunks (async when available)
                for (int kb = k0; kb < k1; kb += 8) {
                    if (kb < nh)               copy_chunk16(&a[kb], hr + kb);
                    else if (kb < 2 * nh)      copy_chunk16(&a[kb], hc + kb - nh);
                    else if (kb < 2 * nh + ne) copy_chunk16(&a[kb], ee + kb - 2 * nh);
                    else { uint4 z = {0, 0, 0, 0}; *(uint4*)&a[kb] = z; }
                }
            } else {
                for (int k = k0; k < k1; ++k) {
                    half_t v = (half_t)0.0f;
                    if      (k < nh)          v = hr[k];
                    else if (k < 2 * nh)      v = hc[k - nh];
                    else if (k < 2 * nh + ne) v = ee[k - 2 * nh];
                    a[k] = v;
                }
            }
        } else {
            for (int kb = k0; kb < k1; kb += 8) {
                uint4 z = {0, 0, 0, 0}; *(uint4*)&a[kb] = z;
            }
        }
    }
    async_lds_wait_all();
    __syncthreads();

    v8f acc[4] = {};
    for (int ks = 0; ks < nk; ++ks) {
        v16h af = *(const v16h*)&At[wave][m * Kpad + ks * 32 + kh * 16];
        #pragma unroll
        for (int nt = 0; nt < 4; ++nt) {
            v16h bf = *(const v16h*)&Wt[(ks * 4 + nt) * 512 + lane * 16];
            acc[nt] = __builtin_amdgcn_wmma_f32_16x16x32_f16(
                false, af, false, bf, (short)0, acc[nt], false, false);
        }
    }

    if (tile * 16 < E) {
        #pragma unroll
        for (int nt = 0; nt < 4; ++nt) {
            const int c = nt * 16 + m;
            const float b = bias[c];
            #pragma unroll
            for (int r2 = 0; r2 < 8; ++r2) {
                const int mm = kh * 8 + r2;
                const int e2 = tile * 16 + mm;
                if (e2 < E)
                    z1[(long)e2 * GHID + c] = (half_t)gelu_exact(acc[nt][r2] + b);
            }
        }
    }
}

// --------------- MLP layer 2 + bias + LayerNorm + residual -----------------
// out = LN(z1 @ W2 + b2) (+ out_old if addRes), rows x 64; writes f32 + f16.
__global__ void __launch_bounds__(128)
k_gemm2_ln(int ROWS, const half_t* __restrict__ z1,
           const half_t* __restrict__ pw, const float* __restrict__ bias,
           const float* __restrict__ gamma, const float* __restrict__ beta,
           float* __restrict__ out32, half_t* __restrict__ out16, int addRes) {
    __shared__ __align__(32) half_t Wt[GHID * GHID];
    __shared__ float Ct[4][16][GHID];
    const int tid = threadIdx.x, wave = tid >> 5, lane = tid & 31;
    const int m = lane & 15, kh = lane >> 4;

    stage_lds16(Wt, pw, GHID * GHID, tid, 128);
    async_lds_wait_all();
    __syncthreads();

    const int tile = blockIdx.x * 4 + wave;
    v8f acc[4] = {};
    #pragma unroll
    for (int ks = 0; ks < 2; ++ks) {
        const int gr = tile * 16 + m;
        v16h af = {};
        if (gr < ROWS) af = *(const v16h*)&z1[(long)gr * GHID + ks * 32 + kh * 16];
        #pragma unroll
        for (int nt = 0; nt < 4; ++nt) {
            v16h bf = *(const v16h*)&Wt[(ks * 4 + nt) * 512 + lane * 16];
            acc[nt] = __builtin_amdgcn_wmma_f32_16x16x32_f16(
                false, af, false, bf, (short)0, acc[nt], false, false);
        }
    }
    #pragma unroll
    for (int nt = 0; nt < 4; ++nt) {
        const int c = nt * 16 + m;
        const float b = bias[c];
        #pragma unroll
        for (int r2 = 0; r2 < 8; ++r2) Ct[wave][kh * 8 + r2][c] = acc[nt][r2] + b;
    }
    __syncthreads();

    if (lane < 16) {
        const int gr = tile * 16 + lane;
        if (gr < ROWS) {
            const float* crow = &Ct[wave][lane][0];
            float mu = 0.0f;
            for (int c = 0; c < GHID; ++c) mu += crow[c];
            mu *= (1.0f / GHID);
            float var = 0.0f;
            for (int c = 0; c < GHID; ++c) { float d = crow[c] - mu; var += d * d; }
            var *= (1.0f / GHID);
            const float inv = rsqrtf(var + G_LN_EPS);
            for (int c = 0; c < GHID; ++c) {
                float v = (crow[c] - mu) * inv * gamma[c] + beta[c];
                const long o = (long)gr * GHID + c;
                if (addRes) v += out32[o];
                out32[o] = v;
                out16[o] = (half_t)v;
            }
        }
    }
}

// ------------------------- edge -> node scatter ----------------------------
__global__ void k_scatter(int E, const int* __restrict__ ecol,
                          const float* __restrict__ e32,
                          float* __restrict__ sadd, float* __restrict__ smax) {
    long i = (long)blockIdx.x * blockDim.x + threadIdx.x;
    if (i >= (long)E * GHID) return;
    const int e = (int)(i >> 6), c = (int)(i & 63);
    const float v = e32[i];
    const long o = (long)ecol[e] * GHID + c;
    atomicAdd(&sadd[o], v);
    atomic_max_f32(&smax[o], v);
}

// ------------------------- node MLP layer 1 --------------------------------
// z1n[n] = gelu( concat(h, s_add, s_max, s_mean, u[batch]) @ W1 + b1 )
__global__ void __launch_bounds__(128)
k_node_gemm1(int N, int nh, int Kpad,
             const half_t* __restrict__ hf16, const float* __restrict__ sadd,
             const float* __restrict__ smax, const float* __restrict__ deg,
             const float* __restrict__ u, const int* __restrict__ batch,
             const half_t* __restrict__ pw, const float* __restrict__ bias,
             half_t* __restrict__ z1n) {
    __shared__ __align__(32) half_t At[4][16 * NG1_MAXK];
    __shared__ __align__(32) half_t Wt[NG1_MAXK * GHID];
    const int tid = threadIdx.x, wave = tid >> 5, lane = tid & 31;
    const int m = lane & 15, kh = lane >> 4;
    const int nk = Kpad >> 5;

    stage_lds16(Wt, pw, Kpad * GHID, tid, 128);

    const int tile = blockIdx.x * 4 + wave;
    {
        const int n = tile * 16 + m;
        half_t* a = &At[wave][m * Kpad];
        const int khalf = Kpad >> 1;
        const int k0 = kh * khalf, k1 = k0 + khalf;
        if (n < N) {
            const float dg = deg[n];
            const float rdeg = 1.0f / fmaxf(dg, 1.0f);
            const int hasin = dg > 0.0f;
            const float ub = u[batch[n]];
            const long nb = (long)n * GHID;
            for (int k = k0; k < k1; ++k) {
                float v = 0.0f;
                if (k < nh) v = (float)hf16[nb + k];
                else {
                    const int kk = k - nh;
                    if      (kk < 64)   v = sadd[nb + kk];
                    else if (kk < 128)  v = hasin ? smax[nb + kk - 64] : 0.0f;
                    else if (kk < 192)  v = sadd[nb + kk - 128] * rdeg;
                    else if (kk == 192) v = ub;
                }
                a[k] = (half_t)v;
            }
        } else {
            for (int kb = k0; kb < k1; kb += 8) {
                uint4 z = {0, 0, 0, 0}; *(uint4*)&a[kb] = z;
            }
        }
    }
    async_lds_wait_all();
    __syncthreads();

    v8f acc[4] = {};
    for (int ks = 0; ks < nk; ++ks) {
        v16h af = *(const v16h*)&At[wave][m * Kpad + ks * 32 + kh * 16];
        #pragma unroll
        for (int nt = 0; nt < 4; ++nt) {
            v16h bf = *(const v16h*)&Wt[(ks * 4 + nt) * 512 + lane * 16];
            acc[nt] = __builtin_amdgcn_wmma_f32_16x16x32_f16(
                false, af, false, bf, (short)0, acc[nt], false, false);
        }
    }

    if (tile * 16 < N) {
        #pragma unroll
        for (int nt = 0; nt < 4; ++nt) {
            const int c = nt * 16 + m;
            const float b = bias[c];
            #pragma unroll
            for (int r2 = 0; r2 < 8; ++r2) {
                const int mm = kh * 8 + r2;
                const int n2 = tile * 16 + mm;
                if (n2 < N)
                    z1n[(long)n2 * GHID + c] = (half_t)gelu_exact(acc[nt][r2] + b);
            }
        }
    }
}

// ------------------------------ pooling ------------------------------------
__global__ void k_pool(int N, const int* __restrict__ batch,
                       const float* __restrict__ h32,
                       float* __restrict__ padd, float* __restrict__ pmax) {
    long i = (long)blockIdx.x * blockDim.x + threadIdx.x;
    if (i >= (long)N * GHID) return;
    const int n = (int)(i >> 6), c = (int)(i & 63);
    const float v = h32[i];
    const long o = (long)batch[n] * GHID + c;
    atomicAdd(&padd[o], v);
    atomic_max_f32(&pmax[o], v);
}

// ------------------------- output head (fp32, G<=16) -----------------------
__global__ void __launch_bounds__(256)
k_out_mlp(int G, const float* __restrict__ padd, const float* __restrict__ pmax,
          const float* __restrict__ gcnt, const float* __restrict__ u,
          const float* __restrict__ w0, const float* __restrict__ b0,
          const float* __restrict__ w1, const float* __restrict__ b1,
          const float* __restrict__ w2, const float* __restrict__ b2,
          const float* __restrict__ w3, const float* __restrict__ b3,
          float* __restrict__ out) {
    __shared__ float IN[16 * 193];
    __shared__ float O1[16 * GHID];
    __shared__ float O2[16 * GHID];
    const int tid = threadIdx.x;
    const int KIN = 3 * GHID + 1; // 193
    for (int i = tid; i < G * KIN; i += 256) {
        const int g = i / KIN, k = i % KIN;
        const float cnt = gcnt[g];
        const float inv = 1.0f / fmaxf(cnt, 1.0f);
        float v;
        if      (k < 64)  v = padd[g * GHID + k];
        else if (k < 128) v = padd[g * GHID + (k - 64)] * inv;
        else if (k < 192) v = (cnt > 0.0f) ? pmax[g * GHID + (k - 128)] : 0.0f;
        else              v = u[g];
        IN[g * KIN + k] = v;
    }
    __syncthreads();
    for (int i = tid; i < G * GHID; i += 256) {
        const int g = i >> 6, c = i & 63;
        float s = b0[c];
        for (int k = 0; k < KIN; ++k) s += IN[g * KIN + k] * w0[k * GHID + c];
        O1[i] = gelu_exact(s);
    }
    __syncthreads();
    for (int i = tid; i < G * GHID; i += 256) {
        const int g = i >> 6, c = i & 63;
        float s = b1[c];
        for (int k = 0; k < GHID; ++k) s += O1[g * GHID + k] * w1[k * GHID + c];
        O2[i] = gelu_exact(s);
    }
    __syncthreads();
    for (int i = tid; i < G * GHID; i += 256) {
        const int g = i >> 6, c = i & 63;
        float s = b2[c];
        for (int k = 0; k < GHID; ++k) s += O2[g * GHID + k] * w2[k * GHID + c];
        O1[i] = gelu_exact(s);     // reuse O1 for stage-3 output
    }
    __syncthreads();
    for (int i = tid; i < G * 2; i += 256) {
        const int g = i >> 1, c = i & 1;
        float s = b3[c];
        for (int k = 0; k < GHID; ++k) s += O1[g * GHID + k] * w3[k * 2 + c];
        float v = softplus_f(s);
        if (c == 1) v *= G_ALPHA;
        out[i] = v;
    }
}

// ============================== launcher ===================================

extern "C" void kernel_launch(void* const* d_in, const int* in_sizes, int n_in,
                              void* d_out, int out_size, void* d_ws, size_t ws_size,
                              hipStream_t stream) {
    const float* x     = (const float*)d_in[0];
    const int*   eidx  = (const int*)d_in[1];
    const float* eattr = (const float*)d_in[2];
    const float* u     = (const float*)d_in[3];
    const int*   batch = (const int*)d_in[4];

    const int N = in_sizes[0] / 7;
    const int E = in_sizes[1] / 2;
    const int G = in_sizes[3];
    const int* erow = eidx;
    const int* ecol = eidx + E;

    // --- parameter leaves (jax pytree order: sorted dict keys, list order) ---
    // per layer: e.l1.b, e.l1.w, e.l2.b, e.l2.w, e.ln.beta, e.ln.g,
    //            n.l1.b, n.l1.w, n.l2.b, n.l2.w, n.ln.beta, n.ln.g
    // out: (b,w) x 4
    long lsz[44];
    {
        int ni = 7, ei = 3, li = 0;
        for (int L = 0; L < 3; ++L) {
            const long eK = 2L * ni + ei, nK = ni + 3L * GHID + 1;
            lsz[li++] = 64; lsz[li++] = eK * 64; lsz[li++] = 64; lsz[li++] = 4096;
            lsz[li++] = 64; lsz[li++] = 64;
            lsz[li++] = 64; lsz[li++] = nK * 64; lsz[li++] = 64; lsz[li++] = 4096;
            lsz[li++] = 64; lsz[li++] = 64;
            ni = GHID; ei = GHID;
        }
        lsz[36] = 64; lsz[37] = 193L * 64; lsz[38] = 64; lsz[39] = 4096;
        lsz[40] = 64; lsz[41] = 4096;      lsz[42] = 2;  lsz[43] = 128;
    }
    const float* leaves[44];
    if (n_in > 6) {
        for (int i = 0; i < 44 && 5 + i < n_in; ++i)
            leaves[i] = (const float*)d_in[5 + i];
    } else {
        const float* P = (const float*)d_in[5];
        long off = 0;
        for (int i = 0; i < 44; ++i) { leaves[i] = P + off; off += lsz[i]; }
    }

    // --- workspace bump allocator ---
    char* wp = (char*)d_ws;
    auto alloc = [&](size_t bytes) -> void* {
        void* p = (void*)wp; wp += (bytes + 255) & ~(size_t)255; return p;
    };
    half_t* hf16 = (half_t*)alloc((size_t)N * GHID * 2);
    half_t* ef16 = (half_t*)alloc((size_t)E * GHID * 2);
    float*  e32  = (float*)alloc((size_t)E * GHID * 4);
    float*  h32  = (float*)alloc((size_t)N * GHID * 4);
    half_t* z1e  = (half_t*)alloc((size_t)E * GHID * 2);
    half_t* z1n  = (half_t*)alloc((size_t)N * GHID * 2);
    float*  sadd = (float*)alloc((size_t)N * GHID * 4);
    float*  smax = (float*)alloc((size_t)N * GHID * 4);
    float*  deg  = (float*)alloc((size_t)N * 4);
    float*  padd = (float*)alloc((size_t)G * GHID * 4);
    float*  pmax = (float*)alloc((size_t)G * GHID * 4);
    float*  gcnt = (float*)alloc((size_t)G * 4);
    half_t* pwE1 = (half_t*)alloc((size_t)EG1_MAXK * GHID * 2);
    half_t* pwE2 = (half_t*)alloc((size_t)GHID * GHID * 2);
    half_t* pwN1 = (half_t*)alloc((size_t)NG1_MAXK * GHID * 2);
    half_t* pwN2 = (half_t*)alloc((size_t)GHID * GHID * 2);

    auto cdiv = [](long a, long b) { return (int)((a + b - 1) / b); };
    const float NEG_INF = -__builtin_huge_valf();

    // --- init ---
    k_fill_f32<<<cdiv((long)N, 256), 256, 0, stream>>>(deg, N, 0.0f);
    k_deg<<<cdiv((long)E, 256), 256, 0, stream>>>(ecol, deg, E);
    k_cvt_f16<<<cdiv((long)N * 7, 256), 256, 0, stream>>>(x, hf16, N, 7, GHID);
    k_cvt_f16<<<cdiv((long)E * 3, 256), 256, 0, stream>>>(eattr, ef16, E, 3, GHID);

    // --- layers ---
    int nh = 7, ne = 3;
    for (int L = 0; L < 3; ++L) {
        const float* const* lv = &leaves[12 * L];
        const int eK = 2 * nh + ne;
        const int eKpad = (eK + 31) & ~31;
        const int nK = nh + 3 * GHID + 1;
        const int nKpad = (nK + 31) & ~31;

        k_pack_w<<<cdiv((long)eKpad * GHID, 256), 256, 0, stream>>>(lv[1], pwE1, eK, eKpad);
        k_pack_w<<<cdiv((long)GHID * GHID, 256), 256, 0, stream>>>(lv[3], pwE2, GHID, GHID);

        const int eTiles = cdiv(E, 16);
        k_edge_gemm1<<<cdiv(eTiles, 4), 128, 0, stream>>>(
            E, nh, ne, eKpad, hf16, ef16, erow, ecol, pwE1, lv[0], z1e);
        k_gemm2_ln<<<cdiv(eTiles, 4), 128, 0, stream>>>(
            E, z1e, pwE2, lv[2], lv[5], lv[4], e32, ef16, L > 0 ? 1 : 0);

        k_fill_f32<<<cdiv((long)N * GHID, 256), 256, 0, stream>>>(sadd, (long)N * GHID, 0.0f);
        k_fill_f32<<<cdiv((long)N * GHID, 256), 256, 0, stream>>>(smax, (long)N * GHID, NEG_INF);
        k_scatter<<<cdiv((long)E * GHID, 256), 256, 0, stream>>>(E, ecol, e32, sadd, smax);

        k_pack_w<<<cdiv((long)nKpad * GHID, 256), 256, 0, stream>>>(lv[7], pwN1, nK, nKpad);
        k_pack_w<<<cdiv((long)GHID * GHID, 256), 256, 0, stream>>>(lv[9], pwN2, GHID, GHID);

        const int nTiles = cdiv(N, 16);
        k_node_gemm1<<<cdiv(nTiles, 4), 128, 0, stream>>>(
            N, nh, nKpad, hf16, sadd, smax, deg, u, batch, pwN1, lv[6], z1n);
        k_gemm2_ln<<<cdiv(nTiles, 4), 128, 0, stream>>>(
            N, z1n, pwN2, lv[8], lv[11], lv[10], h32, hf16, L > 0 ? 1 : 0);

        nh = GHID; ne = GHID;
    }

    // --- pooling + output head ---
    k_fill_f32<<<cdiv((long)G * GHID, 256), 256, 0, stream>>>(padd, (long)G * GHID, 0.0f);
    k_fill_f32<<<cdiv((long)G * GHID, 256), 256, 0, stream>>>(pmax, (long)G * GHID, NEG_INF);
    k_fill_f32<<<cdiv((long)G, 256), 256, 0, stream>>>(gcnt, G, 0.0f);
    k_gcnt<<<cdiv((long)N, 256), 256, 0, stream>>>(batch, gcnt, N);
    k_pool<<<cdiv((long)N * GHID, 256), 256, 0, stream>>>(N, batch, h32, padd, pmax);

    k_out_mlp<<<1, 256, 0, stream>>>(G, padd, pmax, gcnt, u,
                                     leaves[37], leaves[36], leaves[39], leaves[38],
                                     leaves[41], leaves[40], leaves[43], leaves[42],
                                     (float*)d_out);
}
